// TopKSAE_27659589386479
// MI455X (gfx1250) — matrix operand order repaired and checked by reading, not compiled, not run on hardware
//
#include <hip/hip_runtime.h>

// ---------------------------------------------------------------------------
// TopK SAE forward for MI455X (gfx1250, wave32, WMMA + TDM)
// N=8192, D_MODEL=2048, D_DICT=16384, K=64
// ---------------------------------------------------------------------------

#define N_ROWS   8192
#define D_MODEL  2048
#define D_DICT   16384
#define TOPK     64

typedef __attribute__((ext_vector_type(16))) __bf16 v16bf;
typedef __attribute__((ext_vector_type(8)))  __bf16 v8bf;
typedef __attribute__((ext_vector_type(8)))  float  v8f;
typedef __attribute__((ext_vector_type(4)))  unsigned int u32x4;
typedef __attribute__((ext_vector_type(8)))  int i32x8;
typedef __attribute__((ext_vector_type(4)))  int i32x4;

// ----------------------------- helpers -------------------------------------

__device__ __forceinline__ unsigned short f2bf(float f) {
    unsigned int u = __float_as_uint(f);
    unsigned int r = u + 0x7FFFu + ((u >> 16) & 1u);   // round-nearest-even
    return (unsigned short)(r >> 16);
}

// A-fragment (16x32 bf16, MxK): lanes 0-15 hold K {0..7,16..23},
// lanes 16-31 hold K {8..15,24..31}; base points at LDS row + k-step base.
__device__ __forceinline__ v16bf load_fragA(const unsigned short* base, int hi) {
    const v8bf lo = *(const v8bf*)(base + (hi ? 8 : 0));
    const v8bf hh = *(const v8bf*)(base + 16 + (hi ? 8 : 0));
    return __builtin_shufflevector(lo, hh, 0,1,2,3,4,5,6,7,8,9,10,11,12,13,14,15);
}

// B-fragment (32x16 bf16, KxN): lanes 0-15 hold K {0..7,16..23}, lanes 16-31 K {8..15,24..31}.
__device__ __forceinline__ v16bf load_fragB(const unsigned short* base, int hi) {
    const v8bf lo = *(const v8bf*)(base + (hi ? 8 : 0));
    const v8bf hh = *(const v8bf*)(base + 16 + (hi ? 8 : 0));
    return __builtin_shufflevector(lo, hh, 0,1,2,3,4,5,6,7,8,9,10,11,12,13,14,15);
}

// ----------------------------- TDM tile loader ------------------------------
// 2D tile: 128 rows x 64 bf16 (128 B per row), row stride 2048 elements in
// global memory. LDS padding: insert 4 dwords (16 B) after every 32 dwords
// (128 B) -> effective LDS row stride 144 B == LDA(72) shorts, matching the
// fragment loaders below.  D# packing per CDNA5 ISA §8.3/§8.4.

#define BM 128
#define BN 128
#define BK 64
#define LDA 72   // padded LDS stride (ushort units): 64 data + 8 pad

__device__ __forceinline__ void tdm_load_tile(unsigned lds_off,
                                              const unsigned short* gsrc,
                                              unsigned rows_total) {
    unsigned long long ga = (unsigned long long)gsrc;
    u32x4 g0 = { 1u,                                            // count=1, user mode
                 lds_off,                                       // lds_addr
                 (unsigned)(ga & 0xFFFFFFFFu),                  // global_addr[31:0]
                 (unsigned)((ga >> 32) & 0x01FFFFFFu) | (2u << 30) }; // addr[56:32] | type=2
    i32x8 g1 = { (int)((1u << 16)        // data_size = 2 bytes
                     | (1u << 20)        // pad_enable
                     | (4u << 22)        // pad_interval: 32 dwords
                     | (3u << 25)),      // pad_amount:   4 dwords
                 (int)(2048u << 16),     // tensor_dim0[15:0]=2048 in bits[63:48]
                 (int)(rows_total << 16),// tensor_dim0[31:16]=0 ; tensor_dim1[15:0]
                 (int)(64u << 16),       // tensor_dim1[31:16]=0 ; tile_dim0=64
                 (int)128,               // tile_dim1=128 ; tile_dim2=0
                 (int)2048,              // tensor_dim0_stride[31:0]=2048
                 0, 0 };                 // stride hi / dim1_stride = 0
    i32x4 gz4 = { 0, 0, 0, 0 };          // groups 2/3 unused (2-D tensor)
    i32x8 gz8 = { 0, 0, 0, 0, 0, 0, 0, 0 };
    __builtin_amdgcn_tensor_load_to_lds(g0, g1, gz4, gz4, gz8, 0);
}

// ----------------------------- conversion ----------------------------------

__global__ void __launch_bounds__(256) cvt_x_kernel(const float* __restrict__ x,
                                                    const float* __restrict__ bdec,
                                                    unsigned short* __restrict__ Xb) {
    size_t i = (size_t)blockIdx.x * 256 + threadIdx.x;   // < 8192*2048
    int c = (int)(i & (D_MODEL - 1));
    Xb[i] = f2bf(x[i] - bdec[c]);
}

__global__ void __launch_bounds__(256) cvt_w_kernel(const float* __restrict__ w,
                                                    unsigned short* __restrict__ Wb) {
    size_t i = (size_t)blockIdx.x * 256 + threadIdx.x;   // < 16384*2048
    Wb[i] = f2bf(w[i]);
}

__global__ void zero_stats_kernel(float* __restrict__ stat) {
    // stat[0]=residual, stat[1]=l0 count (uint), stat[2..2049]=colS, [2050..4097]=colQ
    for (int i = threadIdx.x; i < 2 + 2 * D_MODEL; i += 256) stat[i] = 0.0f;
}

// ----------------------------- encoder GEMM --------------------------------
// pre[n,d] = relu( sum_k Xc_bf16[n,k] * Wenc_bf16[d,k] + b_enc[d] )
// Tile: BM=128 x BN=128 x BK=64. 256 threads = 8 waves (4x2); each wave owns
// 32 rows x 64 cols = 2x4 16x16 f32 accumulators (64 VGPRs resident).
// Tiles are streamed into double-buffered LDS by the Tensor Data Mover,
// issued by wave 0; completion via TENSORcnt (in-order per wave).

__global__ void __launch_bounds__(256) encoder_gemm_kernel(
        const unsigned short* __restrict__ Xb,
        const unsigned short* __restrict__ Wb,
        const float* __restrict__ benc,
        float* __restrict__ pre) {
    __shared__ unsigned short As[2][BM][LDA];   // 2 x 18 KB
    __shared__ unsigned short Bs[2][BN][LDA];   // 2 x 18 KB

    const int tid  = threadIdx.x;
    const int wave = tid >> 5;
    const int lane = tid & 31;
    const int lr   = lane & 15;
    const int hi   = lane >> 4;
    const int wm   = wave >> 1;   // 0..3  (row block of 32)
    const int wn   = wave & 1;    // 0..1  (col block of 64)

    const size_t rowBase = (size_t)blockIdx.y * BM;
    const size_t colBase = (size_t)blockIdx.x * BN;

    v8f acc[2][4];
    #pragma unroll
    for (int i = 0; i < 2; ++i)
        #pragma unroll
        for (int j = 0; j < 4; ++j)
            acc[i][j] = (v8f)(0.0f);

    const unsigned short* Ag = Xb + rowBase * D_MODEL;
    const unsigned short* Bg = Wb + colBase * D_MODEL;

    const unsigned ldsA0 = (unsigned)(unsigned long long)(void*)&As[0][0][0];
    const unsigned ldsA1 = (unsigned)(unsigned long long)(void*)&As[1][0][0];
    const unsigned ldsB0 = (unsigned)(unsigned long long)(void*)&Bs[0][0][0];
    const unsigned ldsB1 = (unsigned)(unsigned long long)(void*)&Bs[1][0][0];

    if (wave == 0) {                      // preload tile 0 into buffer 0
        tdm_load_tile(ldsA0, Ag, N_ROWS);
        tdm_load_tile(ldsB0, Bg, D_DICT);
    }

    for (int k0 = 0; k0 < D_MODEL; k0 += BK) {
        const int cur = (k0 >> 6) & 1;
        if (wave == 0) {
            if (k0 + BK < D_MODEL) {      // prefetch next tile into other buffer
                tdm_load_tile(cur ? ldsA0 : ldsA1, Ag + k0 + BK, N_ROWS);
                tdm_load_tile(cur ? ldsB0 : ldsB1, Bg + k0 + BK, D_DICT);
                // oldest 2 tensor ops (current tile) done; 2 newest in flight
                __builtin_amdgcn_s_wait_tensorcnt((unsigned short)2);
            } else {
                __builtin_amdgcn_s_wait_tensorcnt((unsigned short)0);
            }
        }
        __syncthreads();                  // current buffer visible to all waves

        #pragma unroll
        for (int ks = 0; ks < BK; ks += 32) {
            v16bf a0 = load_fragA(&As[cur][wm * 32 +      lr][ks], hi);
            v16bf a1 = load_fragA(&As[cur][wm * 32 + 16 + lr][ks], hi);
            v16bf b0 = load_fragB(&Bs[cur][wn * 64 +      lr][ks], hi);
            v16bf b1 = load_fragB(&Bs[cur][wn * 64 + 16 + lr][ks], hi);
            v16bf b2 = load_fragB(&Bs[cur][wn * 64 + 32 + lr][ks], hi);
            v16bf b3 = load_fragB(&Bs[cur][wn * 64 + 48 + lr][ks], hi);

            acc[0][0] = __builtin_amdgcn_wmma_f32_16x16x32_bf16(false, a0, false, b0, (short)0, acc[0][0], false, false);
            acc[0][1] = __builtin_amdgcn_wmma_f32_16x16x32_bf16(false, a0, false, b1, (short)0, acc[0][1], false, false);
            acc[0][2] = __builtin_amdgcn_wmma_f32_16x16x32_bf16(false, a0, false, b2, (short)0, acc[0][2], false, false);
            acc[0][3] = __builtin_amdgcn_wmma_f32_16x16x32_bf16(false, a0, false, b3, (short)0, acc[0][3], false, false);
            acc[1][0] = __builtin_amdgcn_wmma_f32_16x16x32_bf16(false, a1, false, b0, (short)0, acc[1][0], false, false);
            acc[1][1] = __builtin_amdgcn_wmma_f32_16x16x32_bf16(false, a1, false, b1, (short)0, acc[1][1], false, false);
            acc[1][2] = __builtin_amdgcn_wmma_f32_16x16x32_bf16(false, a1, false, b2, (short)0, acc[1][2], false, false);
            acc[1][3] = __builtin_amdgcn_wmma_f32_16x16x32_bf16(false, a1, false, b3, (short)0, acc[1][3], false, false);
        }
        __syncthreads();                  // all reads done before buffer reuse
    }

    // C layout: VGPR v -> M = v + 8*hi, N = lane%16
    #pragma unroll
    for (int bn = 0; bn < 4; ++bn) {
        size_t gc = colBase + wn * 64 + bn * 16 + lr;
        float bias = benc[gc];
        #pragma unroll
        for (int bm = 0; bm < 2; ++bm) {
            #pragma unroll
            for (int v = 0; v < 8; ++v) {
                size_t gr = rowBase + wm * 32 + bm * 16 + hi * 8 + v;
                float val = acc[bm][bn][v] + bias;
                pre[gr * D_DICT + gc] = val > 0.0f ? val : 0.0f;
            }
        }
    }
}

// ----------------------------- top-k select --------------------------------
// One workgroup per row. Relu'd values are >= 0, so float bits are uint-ordered.
// 4-pass radix select of the 64th largest, then rewrite row + emit pairs.

__global__ void __launch_bounds__(256) topk_kernel(float* __restrict__ h,
                                                   int* __restrict__ pairs,
                                                   float* __restrict__ stat) {
    __shared__ unsigned int keys[D_DICT];       // 64 KB (of 320 KB WGP LDS)
    __shared__ unsigned int hist[256];
    __shared__ unsigned int sPrefix, sKRem, sTie, sSlot, sL0;

    const int row = blockIdx.x;
    const int t   = threadIdx.x;
    float* hrow = h + (size_t)row * D_DICT;

    for (int i = t; i < D_DICT; i += 256) keys[i] = __float_as_uint(hrow[i]);
    if (t == 0) { sPrefix = 0u; sKRem = TOPK; sTie = 0u; sSlot = 0u; sL0 = 0u; }
    __syncthreads();

    unsigned int mask = 0u, prefix = 0u;
    for (int shift = 24; shift >= 0; shift -= 8) {
        hist[t] = 0u;
        __syncthreads();
        for (int i = t; i < D_DICT; i += 256) {
            unsigned int kk = keys[i];
            if ((kk & mask) == prefix) atomicAdd(&hist[(kk >> shift) & 255u], 1u);
        }
        __syncthreads();
        if (t == 0) {
            unsigned int need = sKRem, cum = 0u;
            int b = 255;
            for (; b > 0; --b) {
                unsigned int c = hist[b];
                if (cum + c >= need) break;
                cum += c;
            }
            sPrefix |= ((unsigned int)b) << shift;
            sKRem = need - cum;
        }
        __syncthreads();
        prefix = sPrefix;
        mask |= 0xFFu << shift;
    }
    const unsigned int T    = prefix;
    const unsigned int kRem = sKRem;

    for (int i = t; i < D_DICT; i += 256) {
        unsigned int kk = keys[i];
        float outv = 0.0f;
        bool keep = false;
        if (kk > T) keep = true;
        else if (kk == T) { unsigned int p = atomicAdd(&sTie, 1u); keep = (p < kRem); }
        if (keep) {
            outv = __uint_as_float(kk);
            unsigned int s = atomicAdd(&sSlot, 1u);
            pairs[(size_t)row * (2 * TOPK) + 2 * s]     = i;
            pairs[(size_t)row * (2 * TOPK) + 2 * s + 1] = (int)kk;
            if (kk > 0u) atomicAdd(&sL0, 1u);
        }
        hrow[i] = outv;
    }
    __syncthreads();
    if (t == 0) atomicAdd((unsigned int*)stat + 1, sL0);
}

// ----------------------------- W_dec transpose ------------------------------
// Wdec [D_MODEL][D_DICT] -> WdT [D_DICT][D_MODEL] (L2-resident for decode).

__global__ void transpose_wdec_kernel(const float* __restrict__ Wdec,
                                      float* __restrict__ WdT) {
    __shared__ float tile[32][33];
    int tx = threadIdx.x, ty = threadIdx.y;          // 32 x 8
    int gx  = blockIdx.x * 32 + tx;                  // dict idx
    int gy0 = blockIdx.y * 32;                       // model idx
    #pragma unroll
    for (int j = 0; j < 32; j += 8)
        tile[ty + j][tx] = Wdec[(size_t)(gy0 + ty + j) * D_DICT + gx];
    __syncthreads();
    int ox  = blockIdx.y * 32 + tx;                  // model idx
    int oy0 = blockIdx.x * 32;                       // dict idx
    #pragma unroll
    for (int j = 0; j < 32; j += 8)
        WdT[(size_t)(oy0 + ty + j) * D_MODEL + ox] = tile[tx][ty + j];
}

// ----------------------------- sparse decode --------------------------------
// x_hat[n,c] = b_dec[c] + sum_s val_s * WdT[idx_s][c]; fused residual reduce.

__global__ void __launch_bounds__(256) decode_kernel(const float* __restrict__ WdT,
                                                     const float* __restrict__ bdec,
                                                     const int* __restrict__ pairs,
                                                     const float* __restrict__ x,
                                                     float* __restrict__ xhat,
                                                     float* __restrict__ stat) {
    __shared__ int   sidx[TOPK];
    __shared__ float sval[TOPK];
    __shared__ float red[256];
    const int row = blockIdx.x;
    const int t   = threadIdx.x;

    if (t < TOPK) {
        sidx[t] = pairs[(size_t)row * (2 * TOPK) + 2 * t];
        sval[t] = __int_as_float(pairs[(size_t)row * (2 * TOPK) + 2 * t + 1]);
    }
    __syncthreads();

    float acc[8];
    #pragma unroll
    for (int j = 0; j < 8; ++j) acc[j] = bdec[t + 256 * j];

    for (int s = 0; s < TOPK; ++s) {
        const float* wr = WdT + (size_t)sidx[s] * D_MODEL;
        float v = sval[s];
        #pragma unroll
        for (int j = 0; j < 8; ++j) acc[j] += v * wr[t + 256 * j];
    }

    const float* xr = x + (size_t)row * D_MODEL;
    float* outr = xhat + (size_t)row * D_MODEL;
    float r = 0.0f;
    #pragma unroll
    for (int j = 0; j < 8; ++j) {
        outr[t + 256 * j] = acc[j];
        float d = acc[j] - xr[t + 256 * j];
        r += d * d;
    }
    red[t] = r;
    __syncthreads();
    for (int o = 128; o > 0; o >>= 1) {
        if (t < o) red[t] += red[t + o];
        __syncthreads();
    }
    if (t == 0) atomicAdd(&stat[0], red[0]);
}

// ----------------------------- column stats ---------------------------------

__global__ void __launch_bounds__(256) colstats_kernel(const float* __restrict__ x,
                                                       float* __restrict__ stat) {
    int c  = blockIdx.x * 256 + threadIdx.x;   // 0..2047
    int r0 = blockIdx.y * 256;
    float s = 0.0f, q = 0.0f;
    for (int r = r0; r < r0 + 256; ++r) {
        float v = x[(size_t)r * D_MODEL + c];
        s += v;
        q += v * v;
    }
    atomicAdd(&stat[2 + c], s);
    atomicAdd(&stat[2 + D_MODEL + c], q);
}

__global__ void __launch_bounds__(256) finalize_kernel(const float* __restrict__ stat,
                                                       float* __restrict__ oscal) {
    __shared__ float red[256];
    const int t = threadIdx.x;
    const float* colS = stat + 2;
    const float* colQ = stat + 2 + D_MODEL;
    float tv = 0.0f;
    for (int c = t; c < D_MODEL; c += 256) {
        float s = colS[c];
        tv += colQ[c] - s * s * (1.0f / (float)N_ROWS);
    }
    red[t] = tv;
    __syncthreads();
    for (int o = 128; o > 0; o >>= 1) {
        if (t < o) red[t] += red[t + o];
        __syncthreads();
    }
    if (t == 0) {
        float resid = stat[0];
        unsigned int l0c = ((const unsigned int*)stat)[1];
        float tvt = red[0];
        oscal[0] = resid / ((float)N_ROWS * (float)D_MODEL);  // recon_loss
        oscal[1] = (float)l0c / (float)N_ROWS;                // l0
        oscal[2] = 1.0f - resid / (tvt + 1e-8f);              // explained_var
    }
}

// ----------------------------- launch ---------------------------------------

extern "C" void kernel_launch(void* const* d_in, const int* in_sizes, int n_in,
                              void* d_out, int out_size, void* d_ws, size_t ws_size,
                              hipStream_t stream) {
    const float* x    = (const float*)d_in[0];
    const float* Wenc = (const float*)d_in[1];
    const float* benc = (const float*)d_in[2];
    const float* Wdec = (const float*)d_in[3];
    const float* bdec = (const float*)d_in[4];

    float* out   = (float*)d_out;
    float* xhat  = out;
    float* h     = out + (size_t)N_ROWS * D_MODEL;
    float* oscal = out + (size_t)N_ROWS * D_MODEL + (size_t)N_ROWS * D_DICT;

    char* ws = (char*)d_ws;
    unsigned short* Xb  = (unsigned short*)(ws + 0);           //  32 MB
    unsigned short* Wb  = (unsigned short*)(ws + 33554432);    //  64 MB
    float*          WdT = (float*)(ws + 100663296);            // 128 MB
    int*            prs = (int*)(ws + 234881024);              //   4 MB
    float*          st  = (float*)(ws + 239075328);            //  16 KB

    cvt_x_kernel<<<(N_ROWS * D_MODEL) / 256, 256, 0, stream>>>(x, bdec, Xb);
    cvt_w_kernel<<<(D_DICT * D_MODEL) / 256, 256, 0, stream>>>(Wenc, Wb);
    zero_stats_kernel<<<1, 256, 0, stream>>>(st);

    dim3 gemmGrid(D_DICT / BN, N_ROWS / BM);
    encoder_gemm_kernel<<<gemmGrid, 256, 0, stream>>>(Xb, Wb, benc, h);

    transpose_wdec_kernel<<<dim3(D_DICT / 32, D_MODEL / 32), dim3(32, 8), 0, stream>>>(Wdec, WdT);

    topk_kernel<<<N_ROWS, 256, 0, stream>>>(h, prs, st);
    decode_kernel<<<N_ROWS, 256, 0, stream>>>(WdT, bdec, prs, x, xhat, st);
    colstats_kernel<<<dim3(D_MODEL / 256, N_ROWS / 256), 256, 0, stream>>>(x, st);
    finalize_kernel<<<1, 256, 0, stream>>>(st, oscal);
}
